// CrossTransformerDynamicQuery_54271206752580
// MI455X (gfx1250) — matrix operand order repaired and checked by reading, not compiled, not run on hardware
//
#include <hip/hip_runtime.h>

// ---------------- problem constants ----------------
static constexpr int Bc = 4, Qc = 1024, Kc = 1024, Cc = 1024, Hc = 16, CHc = 64;
static constexpr int BQc = Bc * Qc;          // 4096
static constexpr int BKc = Bc * Kc;          // 4096
static constexpr int N3c = 3 * Cc;           // 3072
static constexpr int C2c = 2 * Cc;           // 2048

typedef __attribute__((ext_vector_type(16))) __bf16 v16bf;
typedef __attribute__((ext_vector_type(8)))  float  v8f;

union Frag16 { uint4 q[2]; v16bf v; };       // 8 dwords == one bf16 WMMA operand

// ---------------- small helpers ----------------
__device__ __forceinline__ unsigned short f2bf(float f) {
  unsigned u = __builtin_bit_cast(unsigned, f);
  unsigned r = u + 0x7fffu + ((u >> 16) & 1u);   // round-to-nearest-even
  return (unsigned short)(r >> 16);
}
__device__ __forceinline__ v8f wmma_bf16(v16bf a, v16bf b, v8f c) {
  return __builtin_amdgcn_wmma_f32_16x16x32_bf16(false, a, false, b, (short)0, c,
                                                 false, false);
}
__device__ __forceinline__ v8f zero8() {
  v8f z;
#pragma unroll
  for (int i = 0; i < 8; ++i) z[i] = 0.f;
  return z;
}

// A fragment (16x32 bf16): dwords 0-3 = K{8h..8h+7}, dwords 4-7 = K{16+8h..23+8h},
// row = lane&15.  Source row-major, K contiguous. Two 16B loads, zero unpack ALU.
__device__ __forceinline__ v16bf frag_a(const unsigned short* p, int ld, int lane) {
  const int half = lane >> 4, m = lane & 15;
  const unsigned short* r = p + (size_t)m * ld;
  Frag16 f;
  f.q[0] = *(const uint4*)(r + 8 * half);
  f.q[1] = *(const uint4*)(r + 16 + 8 * half);
  return f.v;
}
// B fragment (32x16 bf16) from K-contiguous (transposed) source:
// col n = lane&15 is a source ROW; dwords 0-3 = K{8h..8h+7}, dwords 4-7 = K{16+8h..23+8h}.
__device__ __forceinline__ v16bf frag_bT(const unsigned short* p, int ld, int lane) {
  const int half = lane >> 4, n = lane & 15;
  const unsigned short* r = p + (size_t)n * ld;
  Frag16 f;
  f.q[0] = *(const uint4*)(r + 8 * half);
  f.q[1] = *(const uint4*)(r + 16 + 8 * half);
  return f.v;
}

// ---------------- elementwise kernels ----------------
__global__ void cast_bf16_kernel(const float* __restrict__ s,
                                 unsigned short* __restrict__ d, size_t n) {
  size_t i = (size_t)blockIdx.x * blockDim.x + threadIdx.x;
  if (i < n) d[i] = f2bf(s[i]);
}

// src [rows, cols] f32 -> dst [cols, rows] bf16   (weight transpose for NT GEMM)
__global__ void transpose_cast_kernel(const float* __restrict__ s,
                                      unsigned short* __restrict__ d,
                                      int rows, int cols) {
  size_t i = (size_t)blockIdx.x * blockDim.x + threadIdx.x;
  if (i < (size_t)rows * cols) {
    int r = (int)(i / cols), c = (int)(i % cols);
    d[(size_t)c * rows + r] = f2bf(s[i]);
  }
}

// LayerNorm over `width`, optional SiLU, bf16 out. One block per row.
__launch_bounds__(256)
__global__ void ln_kernel(const float* __restrict__ x, const float* __restrict__ w,
                          const float* __restrict__ b, unsigned short* __restrict__ out,
                          int width, int do_silu) {
  __shared__ float sA[256], sB[256];
  const int row = blockIdx.x, tid = threadIdx.x;
  const float* xr = x + (size_t)row * width;
  float a = 0.f, q = 0.f;
  for (int i = tid; i < width; i += 256) { float v = xr[i]; a += v; q += v * v; }
  sA[tid] = a; sB[tid] = q; __syncthreads();
  for (int off = 128; off > 0; off >>= 1) {
    if (tid < off) { sA[tid] += sA[tid + off]; sB[tid] += sB[tid + off]; }
    __syncthreads();
  }
  const float mean = sA[0] / width;
  const float var  = sB[0] / width - mean * mean;
  const float rstd = rsqrtf(var + 1e-5f);
  unsigned short* orow = out + (size_t)row * width;
  for (int i = tid; i < width; i += 256) {
    float y = (xr[i] - mean) * rstd * w[i] + b[i];
    if (do_silu) y = y / (1.f + __expf(-y));
    orow[i] = f2bf(y);
  }
}

// GRUCell pointwise: gi/gh are [BQ, 3L] f32, hf [BQ,L]; writes qh f32 + bf16.
__global__ void gru_kernel(const float* __restrict__ gi, const float* __restrict__ gh,
                           const float* __restrict__ hf, float* __restrict__ qh_out,
                           unsigned short* __restrict__ qh_bf) {
  size_t j = (size_t)blockIdx.x * blockDim.x + threadIdx.x;
  if (j >= (size_t)BQc * Cc) return;
  int row = (int)(j >> 10), c = (int)(j & 1023);
  size_t base = (size_t)row * N3c + c;
  float gir = gi[base], giz = gi[base + Cc], gin = gi[base + 2 * Cc];
  float ghr = gh[base], ghz = gh[base + Cc], ghn = gh[base + 2 * Cc];
  float r = 1.f / (1.f + __expf(-(gir + ghr)));
  float z = 1.f / (1.f + __expf(-(giz + ghz)));
  float n = tanhf(gin + r * ghn);
  float h = (1.f - z) * n + z * hf[j];
  qh_out[j] = h;
  qh_bf[j]  = f2bf(h);
}

// ---------------- bf16 WMMA GEMM (NT: A[M,Kd] x Bt[N,Kd]^T), 128x128 tile ----
// 256 threads = 8 waves (4x2); each wave owns 32x64 (2x4 WMMA accumulators).
// All LDS traffic is b128; fragments are 2 x ds_load_b128 each.
__launch_bounds__(256)
__global__ void gemm_bf16_nt(const unsigned short* __restrict__ A,
                             const unsigned short* __restrict__ Bt,
                             const float* __restrict__ bias,
                             const float* __restrict__ resid,
                             float* __restrict__ outF,
                             unsigned short* __restrict__ outB,
                             int M, int N, int Kd) {
  __shared__ __align__(16) unsigned short As[128][40];  // stride 40 shorts = 80B
  __shared__ __align__(16) unsigned short Bs[128][40];
  const int tid = threadIdx.x, lane = tid & 31, wave = tid >> 5;
  const int wm = wave >> 1, wn = wave & 1;
  const int m0 = blockIdx.x * 128, n0 = blockIdx.y * 128;

  v8f acc[2][4];
#pragma unroll
  for (int i = 0; i < 2; ++i)
#pragma unroll
    for (int j = 0; j < 4; ++j) acc[i][j] = zero8();

  for (int k0 = 0; k0 < Kd; k0 += 32) {
    // stage tiles: 128x32 shorts = 512 uint4 each; 2 uint4 per thread per tile
#pragma unroll
    for (int i = 0; i < 2; ++i) {
      int j = tid + i * 256;            // uint4 index
      int r = j >> 2, c = (j & 3) * 8;  // 4 uint4 per 32-short row
      *(uint4*)(&As[r][c]) = *(const uint4*)(A + (size_t)(m0 + r) * Kd + k0 + c);
    }
#pragma unroll
    for (int i = 0; i < 2; ++i) {
      int j = tid + i * 256;
      int r = j >> 2, c = (j & 3) * 8;
      *(uint4*)(&Bs[r][c]) = *(const uint4*)(Bt + (size_t)(n0 + r) * Kd + k0 + c);
    }
    __syncthreads();
    v16bf af0 = frag_a(&As[wm * 32][0], 40, lane);
    v16bf af1 = frag_a(&As[wm * 32 + 16][0], 40, lane);
#pragma unroll
    for (int nt = 0; nt < 4; ++nt) {
      v16bf bf = frag_bT(&Bs[wn * 64 + nt * 16][0], 40, lane);
      acc[0][nt] = wmma_bf16(af0, bf, acc[0][nt]);
      acc[1][nt] = wmma_bf16(af1, bf, acc[1][nt]);
    }
    __syncthreads();
  }

  const int half = lane >> 4, ln = lane & 15;
#pragma unroll
  for (int mi = 0; mi < 2; ++mi)
#pragma unroll
    for (int nt = 0; nt < 4; ++nt)
#pragma unroll
      for (int r = 0; r < 8; ++r) {
        int row = m0 + wm * 32 + mi * 16 + r + 8 * half;
        int col = n0 + wn * 64 + nt * 16 + ln;
        float v = acc[mi][nt][r];
        if (bias)  v += bias[col];
        if (resid) v += resid[(size_t)row * N + col];
        if (outF)  outF[(size_t)row * N + col] = v;
        if (outB)  outB[(size_t)row * N + col] = f2bf(v);
      }
}

// ---------------- fused flash attention (per-head), WMMA ----------------
// grid = (Q/128, B*H); 8 waves, each owns 16 q-rows, flash loop over K in 32s.
// Q and K^T fragments load straight from global (2 x b128); V is staged
// transposed in LDS once per chunk (shared by all waves); P relayout via LDS.
// If `energy` != null, raw (unscaled) scores are streamed out (CA output).
__launch_bounds__(256)
__global__ void attn_kernel(const unsigned short* __restrict__ Qm, int ldq, int qmul, int qadd,
                            const unsigned short* __restrict__ Km, int ldk, int kmul, int kadd,
                            const unsigned short* __restrict__ Vm, int ldv, int vmul, int vadd,
                            float* __restrict__ energy,
                            unsigned short* __restrict__ Om,
                            float scale) {
  __shared__ __align__(16) unsigned short stage[8][16][40]; // per-wave P (C->A frag)
  __shared__ __align__(16) unsigned short Vt[64][40];       // V chunk, transposed
  const int tid = threadIdx.x, lane = tid & 31, wave = tid >> 5;
  const int half = lane >> 4, ln = lane & 15;
  const int bh = blockIdx.y, b = bh >> 4, h = bh & 15;
  const int q0 = blockIdx.x * 128 + wave * 16;

  const unsigned short* qb = Qm + (size_t)(b * Qc + q0) * ldq + h * qmul + qadd;
  const unsigned short* kb = Km + (size_t)(b * Kc) * ldk + h * kmul + kadd;
  const unsigned short* vb = Vm + (size_t)(b * Kc) * ldv + h * vmul + vadd;

  const v16bf a0 = frag_a(qb, ldq, lane);        // head K-dim 0..31
  const v16bf a1 = frag_a(qb + 32, ldq, lane);   // head K-dim 32..63

  float mrun[8], lrun[8], alpha[8];
  v8f o[4];
#pragma unroll
  for (int r = 0; r < 8; ++r) { mrun[r] = -3.0e38f; lrun[r] = 0.f; }
#pragma unroll
  for (int nt = 0; nt < 4; ++nt) o[nt] = zero8();

  for (int kc = 0; kc < Kc; kc += 32) {
    // ---- cooperatively stage V[kc..kc+31][0..63] transposed into Vt[n][k] ----
#pragma unroll
    for (int i = 0; i < 4; ++i) {
      int j = tid + i * 256;            // dword index 0..1023 (32 dwords per k)
      int k = j >> 5, n2 = (j & 31) * 2;
      unsigned d = *(const unsigned*)(vb + (size_t)(kc + k) * ldv + n2);
      Vt[n2][k]     = (unsigned short)(d & 0xffffu);
      Vt[n2 + 1][k] = (unsigned short)(d >> 16);
    }

    // ---- S = Q * K^T : two 16x16 score tiles ----
    v8f s0 = zero8(), s1 = zero8();
    v16bf bt;
    bt = frag_bT(kb + (size_t)(kc)      * ldk + 0,  ldk, lane); s0 = wmma_bf16(a0, bt, s0);
    bt = frag_bT(kb + (size_t)(kc)      * ldk + 32, ldk, lane); s0 = wmma_bf16(a1, bt, s0);
    bt = frag_bT(kb + (size_t)(kc + 16) * ldk + 0,  ldk, lane); s1 = wmma_bf16(a0, bt, s1);
    bt = frag_bT(kb + (size_t)(kc + 16) * ldk + 32, ldk, lane); s1 = wmma_bf16(a1, bt, s1);

    if (energy) {                               // cross-attn: emit UNscaled scores
#pragma unroll
      for (int r = 0; r < 8; ++r) {
        size_t eb = ((size_t)(b * Hc + h) * Qc + (q0 + r + 8 * half)) * (size_t)Kc + kc;
        energy[eb + ln]      = s0[r];
        energy[eb + 16 + ln] = s1[r];
      }
    }

    // ---- online softmax (rows on 16-lane groups; shfl_xor<16 reductions) ----
#pragma unroll
    for (int r = 0; r < 8; ++r) {
      float v0 = s0[r] * scale, v1 = s1[r] * scale;
      float mx = fmaxf(v0, v1);
#pragma unroll
      for (int off = 1; off < 16; off <<= 1) mx = fmaxf(mx, __shfl_xor(mx, off, 32));
      float mn = fmaxf(mrun[r], mx);
      float al = __expf(mrun[r] - mn);
      float p0 = __expf(v0 - mn), p1 = __expf(v1 - mn);
      float rs = p0 + p1;
#pragma unroll
      for (int off = 1; off < 16; off <<= 1) rs += __shfl_xor(rs, off, 32);
      lrun[r] = lrun[r] * al + rs;
      mrun[r] = mn;
      alpha[r] = al;
      stage[wave][r + 8 * half][ln]      = f2bf(p0);
      stage[wave][r + 8 * half][16 + ln] = f2bf(p1);
    }
    __syncthreads();                            // uniform trip count: safe

    // ---- O = diag(alpha)*O + P @ V ----
    v16bf pa = frag_a(&stage[wave][0][0], 40, lane);
#pragma unroll
    for (int nt = 0; nt < 4; ++nt) {
#pragma unroll
      for (int r = 0; r < 8; ++r) o[nt][r] *= alpha[r];
      v16bf bv = frag_bT(&Vt[nt * 16][0], 40, lane);
      o[nt] = wmma_bf16(pa, bv, o[nt]);
    }
    __syncthreads();
  }

#pragma unroll
  for (int nt = 0; nt < 4; ++nt)
#pragma unroll
    for (int r = 0; r < 8; ++r) {
      int row = q0 + r + 8 * half;
      int col = h * CHc + nt * 16 + ln;
      Om[(size_t)(b * Qc + row) * Cc + col] = f2bf(o[nt][r] / lrun[r]);
    }
}

// ---------------- host orchestration ----------------
extern "C" void kernel_launch(void* const* d_in, const int* in_sizes, int n_in,
                              void* d_out, int out_size, void* d_ws, size_t ws_size,
                              hipStream_t stream) {
  (void)in_sizes; (void)n_in; (void)out_size; (void)ws_size;
  const float* queries   = (const float*)d_in[0];
  const float* contexts  = (const float*)d_in[1];
  const float* qhidden   = (const float*)d_in[2];
  const float* sa_norm_w = (const float*)d_in[3];
  const float* sa_norm_b = (const float*)d_in[4];
  const float* sa_qkv_w  = (const float*)d_in[5];
  const float* sa_out_w  = (const float*)d_in[6];
  const float* sa_out_b  = (const float*)d_in[7];
  const float* ca_norm_w = (const float*)d_in[8];
  const float* ca_norm_b = (const float*)d_in[9];
  const float* gru_w_ih  = (const float*)d_in[10];
  const float* gru_w_hh  = (const float*)d_in[11];
  const float* gru_b_ih  = (const float*)d_in[12];
  const float* gru_b_hh  = (const float*)d_in[13];
  const float* ca_k_w    = (const float*)d_in[14];
  const float* ca_v_w    = (const float*)d_in[15];
  const float* ca_out_w  = (const float*)d_in[16];
  const float* ca_out_b  = (const float*)d_in[17];
  const float* ln1_w     = (const float*)d_in[18];
  const float* ln1_b     = (const float*)d_in[19];
  const float* lin1_w    = (const float*)d_in[20];
  const float* ln2_w     = (const float*)d_in[21];
  const float* ln2_b     = (const float*)d_in[22];
  const float* lin2_w    = (const float*)d_in[23];

  float* out0   = (float*)d_out;                             // [B,Q,C]
  float* energy = out0 + (size_t)Bc * Qc * Cc;               // [B,H,Q,K]
  float* qh_out = energy + (size_t)Bc * Hc * Qc * Kc;        // [B,Q,H,CH]

  // ---- workspace layout (bump allocator + two aliased big regions) ----
  char* W = (char*)d_ws;
  size_t off = 0;
  auto alloc = [&](size_t bytes) -> void* {
    void* p = W + off;
    off += (bytes + 255) & ~(size_t)255;
    return p;
  };
  // bf16 weights, all stored TRANSPOSED for the NT GEMM: Bt[N, Kd]
  unsigned short* wb_qkv   = (unsigned short*)alloc((size_t)Cc * N3c * 2);  // [3072,1024]
  unsigned short* wb_saout = (unsigned short*)alloc((size_t)Cc * Cc * 2);
  unsigned short* wb_wih   = (unsigned short*)alloc((size_t)Cc * N3c * 2);  // already [3L,C]
  unsigned short* wb_whh   = (unsigned short*)alloc((size_t)Cc * N3c * 2);
  unsigned short* wb_cak   = (unsigned short*)alloc((size_t)Cc * Cc * 2);
  unsigned short* wb_cav   = (unsigned short*)alloc((size_t)Cc * Cc * 2);
  unsigned short* wb_caout = (unsigned short*)alloc((size_t)Cc * Cc * 2);
  unsigned short* wb_lin1  = (unsigned short*)alloc((size_t)Cc * C2c * 2);  // [2048,1024]
  unsigned short* wb_lin2  = (unsigned short*)alloc((size_t)C2c * Cc * 2);  // [1024,2048]
  // persistent activations
  unsigned short* ctx_bf = (unsigned short*)alloc((size_t)BKc * Cc * 2);
  unsigned short* h_bf   = (unsigned short*)alloc((size_t)BQc * Cc * 2);
  float*          x_f32  = (float*)alloc((size_t)BQc * Cc * 4);
  float*          x2_f32 = (float*)alloc((size_t)BQc * Cc * 4);
  unsigned short* xn2_bf = (unsigned short*)alloc((size_t)BQc * Cc * 2);
  unsigned short* qh_bf  = (unsigned short*)alloc((size_t)BQc * Cc * 2);
  unsigned short* kk_bf  = (unsigned short*)alloc((size_t)BKc * Cc * 2);
  unsigned short* vv_bf  = (unsigned short*)alloc((size_t)BKc * Cc * 2);
  unsigned short* oca_bf = (unsigned short*)alloc((size_t)BQc * Cc * 2);
  unsigned short* s1_bf  = (unsigned short*)alloc((size_t)BQc * Cc * 2);
  // bigA: phase-A {xn, qkv, o_sa} bf16, later reused as gi f32
  char* bigA = (char*)alloc((size_t)BQc * N3c * 4);
  unsigned short* xn_bf  = (unsigned short*)bigA;
  unsigned short* qkv_bf = (unsigned short*)(bigA + (size_t)BQc * Cc * 2);
  unsigned short* osa_bf = (unsigned short*)(bigA + (size_t)BQc * Cc * 2 + (size_t)BQc * N3c * 2);
  float* gi = (float*)bigA;
  // bigB: gh f32, later reused as {m1 f32, s2 bf16}
  char* bigB = (char*)alloc((size_t)BQc * N3c * 4);
  float* gh = (float*)bigB;
  float* m1 = (float*)bigB;
  unsigned short* s2_bf = (unsigned short*)(bigB + (size_t)BQc * C2c * 4);

  const float scale = 1.0f / 32.0f;   // 1/sqrt(C)

  auto cast = [&](const float* s, unsigned short* d, size_t n) {
    cast_bf16_kernel<<<dim3((unsigned)((n + 255) / 256)), dim3(256), 0, stream>>>(s, d, n);
  };
  auto tcast = [&](const float* s, unsigned short* d, int rows, int cols) {
    transpose_cast_kernel<<<dim3((unsigned)(((size_t)rows * cols) / 256)), dim3(256), 0,
                            stream>>>(s, d, rows, cols);
  };
  auto gemm = [&](const unsigned short* A, const unsigned short* Bt, const float* bias,
                  const float* resid, float* oF, unsigned short* oB, int M, int N, int Kd) {
    gemm_bf16_nt<<<dim3(M / 128, N / 128), dim3(256), 0, stream>>>(A, Bt, bias, resid,
                                                                   oF, oB, M, N, Kd);
  };

  // ---- weight/input casts (weights transposed to [N, Kd]) ----
  tcast(sa_qkv_w, wb_qkv, Cc, N3c);
  tcast(sa_out_w, wb_saout, Cc, Cc);
  cast(gru_w_ih, wb_wih, (size_t)N3c * Cc);    // W_ih already [3L, C] == Bt
  cast(gru_w_hh, wb_whh, (size_t)N3c * Cc);
  tcast(ca_k_w, wb_cak, Cc, Cc);
  tcast(ca_v_w, wb_cav, Cc, Cc);
  tcast(ca_out_w, wb_caout, Cc, Cc);
  tcast(lin1_w, wb_lin1, Cc, C2c);
  tcast(lin2_w, wb_lin2, C2c, Cc);
  cast(contexts, ctx_bf, (size_t)BKc * Cc);
  cast(qhidden, h_bf, (size_t)BQc * Cc);

  // ---- phase A: self-attention block ----
  ln_kernel<<<dim3(BQc), dim3(256), 0, stream>>>(queries, sa_norm_w, sa_norm_b, xn_bf, Cc, 0);
  gemm(xn_bf, wb_qkv, nullptr, nullptr, nullptr, qkv_bf, BQc, N3c, Cc);
  attn_kernel<<<dim3(Qc / 128, Bc * Hc), dim3(256), 0, stream>>>(
      qkv_bf, N3c, 192, 0,            // q slice
      qkv_bf, N3c, 192, 64,           // k slice
      qkv_bf, N3c, 192, 128,          // v slice
      nullptr, osa_bf, scale);
  gemm(osa_bf, wb_saout, sa_out_b, queries, x_f32, nullptr, BQc, Cc, Cc);

  // ---- phase B: GRU query update (qh is output #3) ----
  ln_kernel<<<dim3(BQc), dim3(256), 0, stream>>>(x_f32, ca_norm_w, ca_norm_b, xn2_bf, Cc, 0);
  gemm(xn2_bf, wb_wih, gru_b_ih, nullptr, gi, nullptr, BQc, N3c, Cc);   // reuses bigA
  gemm(h_bf,   wb_whh, gru_b_hh, nullptr, gh, nullptr, BQc, N3c, Cc);
  gru_kernel<<<dim3((BQc * Cc) / 256), dim3(256), 0, stream>>>(gi, gh, qhidden, qh_out, qh_bf);

  // ---- phase C: cross-attention (energy is output #2, written in-kernel) ----
  gemm(ctx_bf, wb_cak, nullptr, nullptr, nullptr, kk_bf, BKc, Cc, Cc);
  gemm(ctx_bf, wb_cav, nullptr, nullptr, nullptr, vv_bf, BKc, Cc, Cc);
  attn_kernel<<<dim3(Qc / 128, Bc * Hc), dim3(256), 0, stream>>>(
      qh_bf, Cc, CHc, 0,
      kk_bf, Cc, CHc, 0,
      vv_bf, Cc, CHc, 0,
      energy, oca_bf, scale);
  gemm(oca_bf, wb_caout, ca_out_b, x_f32, x2_f32, nullptr, BQc, Cc, Cc);

  // ---- phase D: MLP with residual into final output ----
  ln_kernel<<<dim3(BQc), dim3(256), 0, stream>>>(x2_f32, ln1_w, ln1_b, s1_bf, Cc, 1);
  gemm(s1_bf, wb_lin1, nullptr, nullptr, m1, nullptr, BQc, C2c, Cc);      // reuses bigB
  ln_kernel<<<dim3(BQc), dim3(256), 0, stream>>>(m1, ln2_w, ln2_b, s2_bf, C2c, 1);
  gemm(s2_bf, wb_lin2, nullptr, x2_f32, out0, nullptr, BQc, Cc, C2c);
}
